// SelfAttention_24094766530688
// MI455X (gfx1250) — matrix operand order repaired and checked by reading, not compiled, not run on hardware
//
#include <hip/hip_runtime.h>

// Problem constants (from reference): B=4, C=512, H=W=64 -> N=4096, D=64
#define NPIX  4096
#define CCH   512
#define DCH   64
#define BATCH 4

typedef __attribute__((ext_vector_type(16))) _Float16 v16h;
typedef __attribute__((ext_vector_type(8)))  _Float16 v8h;
typedef __attribute__((ext_vector_type(8)))  float    v8f;
typedef __attribute__((ext_vector_type(4)))  float    v4f;
typedef __attribute__((ext_vector_type(4)))  int      v4i;

union V16 {
    v16h v;
    v8h  h[2];
    _Float16 e[16];
};

#if __has_builtin(__builtin_amdgcn_global_load_async_to_lds_b128) && \
    __has_builtin(__builtin_amdgcn_s_wait_asynccnt)
#define HAVE_ASYNC 1
#else
#define HAVE_ASYNC 0
#endif

#if HAVE_ASYNC
// signature (from hipcc diagnostic): (int4 global*, int4 lds*, imm, imm)
#define ASYNC_B128(gp, lp)                                                   \
    __builtin_amdgcn_global_load_async_to_lds_b128(                          \
        (__attribute__((address_space(1))) v4i*)(gp),                        \
        (__attribute__((address_space(3))) v4i*)(lp), 0, 0)
#endif

// ---------------------------------------------------------------------------
// Kernel 0: convert weights to f16. Wq/Wk/Wv (D,C) f32 -> transposed (C,D) f16
// so that WMMA B-fragments (lane = N = d, elements = K = c) are contiguous.
// Wo (C,D) f32 -> f16 straight copy (used as A matrix, rows contiguous in d).
// ---------------------------------------------------------------------------
__global__ void prep_weights(const float* __restrict__ Wq,
                             const float* __restrict__ Wk,
                             const float* __restrict__ Wv,
                             const float* __restrict__ Wo,
                             _Float16* __restrict__ WqT,
                             _Float16* __restrict__ WkT,
                             _Float16* __restrict__ WvT,
                             _Float16* __restrict__ Wo16) {
    int i = blockIdx.x * blockDim.x + threadIdx.x;
    if (i < DCH * CCH) {
        int d = i / CCH;
        int c = i % CCH;
        WqT[c * DCH + d] = (_Float16)Wq[i];
        WkT[c * DCH + d] = (_Float16)Wk[i];
        WvT[c * DCH + d] = (_Float16)Wv[i];
        Wo16[i]          = (_Float16)Wo[i];   // same element count C*D
    }
}

// ---------------------------------------------------------------------------
// Kernel 1: QKV projection.  q[n][d] = sum_c Wq[d][c] * x[c][n] + bq[d]
// Block = 8 waves covering 128 pixels. The x tile (32c x 128n f32, 16 KB) is
// double-buffered in LDS via async global->LDS copies and shared by all
// waves. Each wave: 16 pixels x all D=64 for q,k,v; 12 wmma per c-step.
// Outputs: Q (N,D) f16 row-major, Kt (D,N) f16 (transposed), V (N,D) f16.
// ---------------------------------------------------------------------------
__global__ void __launch_bounds__(256)
qkv_proj(const float* __restrict__ x,
         const _Float16* __restrict__ WqT,
         const _Float16* __restrict__ WkT,
         const _Float16* __restrict__ WvT,
         const float* __restrict__ bq,
         const float* __restrict__ bk,
         const float* __restrict__ bv,
         _Float16* __restrict__ Q,
         _Float16* __restrict__ Kt,
         _Float16* __restrict__ V) {
    __shared__ __align__(128) float Xtile[2][32 * 128];

    int tid  = threadIdx.x;
    int lane = tid & 31;
    int wq   = tid >> 5;
    int blk  = blockIdx.x;           // 0..127
    int b    = blk >> 5;             // 32 blocks per batch
    int nblk = (blk & 31) * 128;     // block pixel base
    int n0   = nblk + wq * 16;       // wave pixel base

    const float* xb = x + (size_t)b * CCH * NPIX;

    int ml = lane & 15;
    int hi = lane >> 4;
    int Kb = hi * 8;

    // cooperative x-tile load: thread -> row c = tid>>3, 16-float chunk
    int xr = tid >> 3;
    int xc = (tid & 7) * 16;

    auto issue = [&](int buf, int c0) {
#if HAVE_ASYNC
#pragma unroll
        for (int j = 0; j < 4; ++j)
            ASYNC_B128(xb + (size_t)(c0 + xr) * NPIX + nblk + xc + 4 * j,
                       &Xtile[buf][xr * 128 + xc + 4 * j]);
#else
#pragma unroll
        for (int j = 0; j < 4; ++j)
            *(v4f*)&Xtile[buf][xr * 128 + xc + 4 * j] =
                *(const v4f*)(xb + (size_t)(c0 + xr) * NPIX + nblk + xc + 4 * j);
#endif
    };

    v8f acc[3][4] = {};     // {q,k,v} x 4 d-tiles

    issue(0, 0);
    for (int it = 0; it < CCH / 32; ++it) {
        int buf = it & 1;
        int c0  = it * 32;
        if (it + 1 < CCH / 32) {
            issue(buf ^ 1, c0 + 32);
#if HAVE_ASYNC
            __builtin_amdgcn_s_wait_asynccnt(4);   // current tile landed
#endif
        } else {
#if HAVE_ASYNC
            __builtin_amdgcn_s_wait_asynccnt(0);
#endif
        }
        __syncthreads();

        // A fragment from LDS: a.e[i] = x[c0+K][n0+ml]
        V16 a;
        int nl = wq * 16 + ml;
#pragma unroll
        for (int i = 0; i < 8; ++i) {
            a.e[i]     = (_Float16)Xtile[buf][(Kb + i) * 128 + nl];
            a.e[i + 8] = (_Float16)Xtile[buf][(16 + Kb + i) * 128 + nl];
        }
        // B-frag: column N = d = 16t+ml, elements over K = c (table order)
#pragma unroll
        for (int t = 0; t < 4; ++t) {
            V16 bq_, bk_, bv_;
#pragma unroll
            for (int i = 0; i < 8; ++i) {
                size_t r0 = (size_t)(c0 + Kb + i) * DCH + 16 * t + ml;
                size_t r1 = (size_t)(c0 + 16 + Kb + i) * DCH + 16 * t + ml;
                bq_.e[i] = WqT[r0]; bq_.e[i + 8] = WqT[r1];
                bk_.e[i] = WkT[r0]; bk_.e[i + 8] = WkT[r1];
                bv_.e[i] = WvT[r0]; bv_.e[i + 8] = WvT[r1];
            }
            v16h bqf = bq_.v;
            acc[0][t] = __builtin_amdgcn_wmma_f32_16x16x32_f16(
                false, a.v, false, bqf, (short)0, acc[0][t], false, false);
            v16h bkf = bk_.v;
            acc[1][t] = __builtin_amdgcn_wmma_f32_16x16x32_f16(
                false, a.v, false, bkf, (short)0, acc[1][t], false, false);
            v16h bvf = bv_.v;
            acc[2][t] = __builtin_amdgcn_wmma_f32_16x16x32_f16(
                false, a.v, false, bvf, (short)0, acc[2][t], false, false);
        }
        __syncthreads();   // before next issue overwrites buf^1
    }

    // C/D layout: element (M = r + 8*hi, N = ml); here M = pixel row, N = d
    size_t base = (size_t)b * NPIX * DCH;
#pragma unroll
    for (int t = 0; t < 4; ++t) {
#pragma unroll
        for (int r = 0; r < 8; ++r) {
            int Mrow = r + 8 * hi;
            int n    = n0 + Mrow;
            int d    = 16 * t + ml;
            Q[base + (size_t)n * DCH + d]   = (_Float16)(acc[0][t][r] + bq[d]);
            Kt[base + (size_t)d * NPIX + n] = (_Float16)(acc[1][t][r] + bk[d]);
            V[base + (size_t)n * DCH + d]   = (_Float16)(acc[2][t][r] + bv[d]);
        }
    }
}

// ---------------------------------------------------------------------------
// Kernel 2: flash attention. Block = 8 waves x 16 query rows = 128 queries.
// Key/value tiles of 32 (Kt: 64d x 32m, V: 32m x 64d, 4 KB each) are
// double-buffered in LDS via async copies and shared by all 8 waves.
// Per tile per wave: 4 wmma for S = Q K^T, online softmax in registers,
// P staged through a private LDS slice (C-layout -> A-layout transpose),
// 4 wmma for acc += P V. Scores never touch global memory.
// Output attT (D,N) f16 so the out-proj B-fragments are contiguous.
// ---------------------------------------------------------------------------
__global__ void __launch_bounds__(256)
flash_attn(const _Float16* __restrict__ Q,
           const _Float16* __restrict__ Kt,
           const _Float16* __restrict__ V,
           _Float16* __restrict__ attT) {
    __shared__ __align__(128) _Float16 Ktile[2][64 * 32];  // [d][m]
    __shared__ __align__(128) _Float16 Vtile[2][32 * 64];  // [m][d]
    __shared__ __align__(64)  _Float16 Plds[8][16 * 32];

    int tid  = threadIdx.x;
    int lane = tid & 31;
    int wq   = tid >> 5;
    int blk  = blockIdx.x;           // 0..127
    int b    = blk >> 5;             // 32 blocks per batch
    int n0   = (blk & 31) * 128 + wq * 16;

    int ml = lane & 15;
    int hi = lane >> 4;
    int Kb = hi * 8;

    size_t base = (size_t)b * NPIX * DCH;
    const _Float16* Qb  = Q + base;
    const _Float16* Ktb = Kt + base;
    const _Float16* Vb  = V + base;

    // cooperative tile-load assignments (16 B per thread per tensor)
    int kd = tid >> 2, kq = (tid & 3) * 8;   // Kt: row d, 8-half chunk
    int vm = tid >> 3, vo = (tid & 7) * 8;   // V : row m, 8-half chunk

    auto issue = [&](int buf, int m0) {
#if HAVE_ASYNC
        ASYNC_B128(Ktb + (size_t)kd * NPIX + m0 + kq, &Ktile[buf][kd * 32 + kq]);
        ASYNC_B128(Vb + (size_t)(m0 + vm) * DCH + vo, &Vtile[buf][vm * 64 + vo]);
#else
        *(v8h*)&Ktile[buf][kd * 32 + kq] =
            *(const v8h*)(Ktb + (size_t)kd * NPIX + m0 + kq);
        *(v8h*)&Vtile[buf][vm * 64 + vo] =
            *(const v8h*)(Vb + (size_t)(m0 + vm) * DCH + vo);
#endif
    };

    // Q A-fragments for the two 32-wide d chunks
    V16 qa[2];
#pragma unroll
    for (int cc = 0; cc < 2; ++cc) {
        const _Float16* qrow = Qb + (size_t)(n0 + ml) * DCH + 32 * cc;
        qa[cc].h[0] = *(const v8h*)(qrow + Kb);
        qa[cc].h[1] = *(const v8h*)(qrow + Kb + 16);
    }

    float mrun[8], lrun[8];
#pragma unroll
    for (int r = 0; r < 8; ++r) { mrun[r] = -1.0e30f; lrun[r] = 0.0f; }
    v8f acc[4] = {};

    _Float16* P = Plds[wq];

    issue(0, 0);
    for (int it = 0; it < NPIX / 32; ++it) {
        int buf = it & 1;
        if (it + 1 < NPIX / 32) {
            issue(buf ^ 1, (it + 1) * 32);
#if HAVE_ASYNC
            __builtin_amdgcn_s_wait_asynccnt(2);   // current tile landed
#endif
        } else {
#if HAVE_ASYNC
            __builtin_amdgcn_s_wait_asynccnt(0);
#endif
        }
        __syncthreads();

        // ---- scores: S[half] is a 16x16 f32 tile over keys m0+16*half ----
        v8f s[2] = {};
#pragma unroll
        for (int half = 0; half < 2; ++half) {
#pragma unroll
            for (int cc = 0; cc < 2; ++cc) {
                V16 bk_;
#pragma unroll
                for (int i = 0; i < 8; ++i) {
                    bk_.e[i]     = Ktile[buf][(32 * cc + Kb + i) * 32 + 16 * half + ml];
                    bk_.e[i + 8] = Ktile[buf][(32 * cc + 16 + Kb + i) * 32 + 16 * half + ml];
                }
                v16h bk = bk_.v;
                s[half] = __builtin_amdgcn_wmma_f32_16x16x32_f16(
                    false, qa[cc].v, false, bk, (short)0, s[half], false, false);
            }
        }

        // ---- online softmax per query row (row M = r + 8*hi spans the
        //      16 lanes of one half; reduce with xor masks 1,2,4,8) ----
#pragma unroll
        for (int r = 0; r < 8; ++r) {
            float t = fmaxf(s[0][r], s[1][r]);
            t = fmaxf(t, __shfl_xor(t, 1, 32));
            t = fmaxf(t, __shfl_xor(t, 2, 32));
            t = fmaxf(t, __shfl_xor(t, 4, 32));
            t = fmaxf(t, __shfl_xor(t, 8, 32));
            float nm   = fmaxf(mrun[r], t);
            float corr = __expf(mrun[r] - nm);
            mrun[r]    = nm;
            float p0 = __expf(s[0][r] - nm);
            float p1 = __expf(s[1][r] - nm);
            float rs = p0 + p1;
            rs += __shfl_xor(rs, 1, 32);
            rs += __shfl_xor(rs, 2, 32);
            rs += __shfl_xor(rs, 4, 32);
            rs += __shfl_xor(rs, 8, 32);
            lrun[r] = lrun[r] * corr + rs;
#pragma unroll
            for (int t4 = 0; t4 < 4; ++t4) acc[t4][r] *= corr;
            int Mrow = r + 8 * hi;
            P[Mrow * 32 + ml]      = (_Float16)p0;   // key col m0+ml
            P[Mrow * 32 + 16 + ml] = (_Float16)p1;   // key col m0+16+ml
        }

        // ---- P (16x32) as A-fragment, V tile (32x64) as B-fragments ----
        V16 pa;
        pa.h[0] = *(const v8h*)(P + ml * 32 + Kb);
        pa.h[1] = *(const v8h*)(P + ml * 32 + Kb + 16);
#pragma unroll
        for (int t4 = 0; t4 < 4; ++t4) {
            V16 bv_;
#pragma unroll
            for (int i = 0; i < 8; ++i) {
                bv_.e[i]     = Vtile[buf][(Kb + i) * 64 + 16 * t4 + ml];
                bv_.e[i + 8] = Vtile[buf][(16 + Kb + i) * 64 + 16 * t4 + ml];
            }
            v16h bv = bv_.v;
            acc[t4] = __builtin_amdgcn_wmma_f32_16x16x32_f16(
                false, pa.v, false, bv, (short)0, acc[t4], false, false);
        }
        __syncthreads();   // before next issue overwrites buf^1
    }

    // normalize and store transposed (d-major) for the output projection
#pragma unroll
    for (int t4 = 0; t4 < 4; ++t4) {
#pragma unroll
        for (int r = 0; r < 8; ++r) {
            int Mrow = r + 8 * hi;
            int d    = 16 * t4 + ml;
            attT[base + (size_t)d * NPIX + n0 + Mrow] =
                (_Float16)(acc[t4][r] / lrun[r]);
        }
    }
}

// ---------------------------------------------------------------------------
// Kernel 3: output projection + residual.
// out[c][n] = gamma * (sum_d Wo[c][d]*att[n][d] + bo[c]) + x[c][n]
// A = Wo tile (16c x 64d), B = attT tile (64d x 16n). One wave: 16c x 64n.
// Weights and attT are L2-resident; direct global B-fragments.
// ---------------------------------------------------------------------------
__global__ void __launch_bounds__(256)
out_proj(const _Float16* __restrict__ Wo16,
         const _Float16* __restrict__ attT,
         const float* __restrict__ bo,
         const float* __restrict__ gamma,
         const float* __restrict__ x,
         float* __restrict__ out) {
    int lane = threadIdx.x & 31;
    int wave = (blockIdx.x * blockDim.x + threadIdx.x) >> 5;  // 0..8191
    int b    = wave >> 11;                                    // 2048 waves/batch
    int rem  = wave & 2047;
    int c0   = (rem >> 6) << 4;   // 32 c-tiles of 16
    int n0   = (rem & 63) << 6;   // 64 n-groups of 64

    int ml = lane & 15;
    int hi = lane >> 4;
    int Kb = hi * 8;

    // A fragments from Wo16 (row c0+ml, K = d)
    V16 wa[2];
#pragma unroll
    for (int cc = 0; cc < 2; ++cc) {
        const _Float16* wrow = Wo16 + (size_t)(c0 + ml) * DCH + 32 * cc;
        wa[cc].h[0] = *(const v8h*)(wrow + Kb);
        wa[cc].h[1] = *(const v8h*)(wrow + Kb + 16);
    }

    size_t abase = (size_t)b * NPIX * DCH;  // attT base (D,N) halfs
    v8f acc[4] = {};
#pragma unroll
    for (int nt = 0; nt < 4; ++nt) {
#pragma unroll
        for (int cc = 0; cc < 2; ++cc) {
            V16 bf_;
#pragma unroll
            for (int i = 0; i < 8; ++i) {
                bf_.e[i]     = attT[abase + (size_t)(32 * cc + Kb + i) * NPIX + n0 + 16 * nt + ml];
                bf_.e[i + 8] = attT[abase + (size_t)(32 * cc + 16 + Kb + i) * NPIX + n0 + 16 * nt + ml];
            }
            v16h bf = bf_.v;
            acc[nt] = __builtin_amdgcn_wmma_f32_16x16x32_f16(
                false, wa[cc].v, false, bf, (short)0, acc[nt], false, false);
        }
    }

    float g = gamma[0];
    size_t xbase = (size_t)b * CCH * NPIX;
#pragma unroll
    for (int nt = 0; nt < 4; ++nt) {
#pragma unroll
        for (int r = 0; r < 8; ++r) {
            int c = c0 + r + 8 * hi;
            int n = n0 + 16 * nt + ml;
            size_t idx = xbase + (size_t)c * NPIX + n;
            out[idx] = g * (acc[nt][r] + bo[c]) + x[idx];
        }
    }
}

// ---------------------------------------------------------------------------
extern "C" void kernel_launch(void* const* d_in, const int* in_sizes, int n_in,
                              void* d_out, int out_size, void* d_ws, size_t ws_size,
                              hipStream_t stream) {
    const float* x     = (const float*)d_in[0];
    const float* Wq    = (const float*)d_in[1];
    const float* bq    = (const float*)d_in[2];
    const float* Wk    = (const float*)d_in[3];
    const float* bk    = (const float*)d_in[4];
    const float* Wv    = (const float*)d_in[5];
    const float* bv    = (const float*)d_in[6];
    const float* Wo    = (const float*)d_in[7];
    const float* bo    = (const float*)d_in[8];
    const float* gamma = (const float*)d_in[9];
    float* out = (float*)d_out;

    // Workspace layout (f16):
    const size_t WSZ = (size_t)DCH * CCH;          // 32768 per weight
    const size_t TSZ = (size_t)BATCH * NPIX * DCH; // 1048576 per tensor
    _Float16* WqT  = (_Float16*)d_ws;
    _Float16* WkT  = WqT + WSZ;
    _Float16* WvT  = WkT + WSZ;
    _Float16* Wo16 = WvT + WSZ;
    _Float16* Qh   = Wo16 + WSZ;
    _Float16* Kth  = Qh + TSZ;
    _Float16* Vh   = Kth + TSZ;
    _Float16* attT = Vh + TSZ;
    // total: 4*32768 + 4*1048576 halfs = ~8.65 MB

    prep_weights<<<(int)((WSZ + 255) / 256), 256, 0, stream>>>(
        Wq, Wk, Wv, Wo, WqT, WkT, WvT, Wo16);

    // 128 blocks x 8 waves; each block 128 pixels
    qkv_proj<<<128, 256, 0, stream>>>(x, WqT, WkT, WvT, bq, bk, bv,
                                      Qh, Kth, Vh);

    // 128 blocks x 8 waves; each block 128 queries
    flash_attn<<<128, 256, 0, stream>>>(Qh, Kth, Vh, attT);

    // 8192 waves (B * C/16 * N/64), 8 per block
    out_proj<<<1024, 256, 0, stream>>>(Wo16, attT, bo, gamma, x, out);
}
